// MultiheadAttentionWrapper_86517821215467
// MI455X (gfx1250) — matrix operand order repaired and checked by reading, not compiled, not run on hardware
//
#include <hip/hip_runtime.h>
#include <hip/hip_bf16.h>

// ---------------------------------------------------------------------------
// Projected MHA for MI455X (gfx1250, wave32, WMMA bf16 16x16x32, f32 accum)
//   q = relu(query @ Wq^T + bq) * hd^-0.5 ; k = relu(key @ Wk^T + bk)
//   out = softmax(q k^T) v   per (b, h)
// ---------------------------------------------------------------------------

constexpr int B_  = 4;
constexpr int T_  = 2048;
constexpr int S_  = 2048;
constexpr int D_  = 1024;
constexpr int H_  = 16;
constexpr int HID_ = 1024;
constexpr int DV_ = 1024;
constexpr int HDI = 64;   // per-head q/k/v dim

typedef __attribute__((ext_vector_type(16))) __bf16 v16bf;
typedef __attribute__((ext_vector_type(8)))  float  v8f;

union Frag {
    v16bf v;
    uint4 u[2];
};

// ---------------------------------------------------------------------------
// Kernel 1: Y[m, n] = relu(X[m,:] . W[n,:] + bias[n]) * scale   (bf16 out)
// One wave computes a 16x64 tile (4 N-tiles), K-loop step 32.
// A layout: lane(0-15)=row, khalf 0 ; lane(16-31)=row, khalf 16, K = khalf + e contiguous
// B layout: lane%16 = column, K = khalf(0/16) + e contiguous
// ---------------------------------------------------------------------------
__global__ __launch_bounds__(32) void proj_relu_kernel(
    const float* __restrict__ X, const float* __restrict__ W,
    const float* __restrict__ bias, __hip_bfloat16* __restrict__ Yo,
    float scale)
{
    __bf16* Y = (__bf16*)Yo;
    const int lane = threadIdx.x;
    const int g    = lane >> 4;        // lane half
    const int ln   = lane & 15;
    const int khA  = g ? 8 : 0;
    const int khB  = g ? 16 : 0;
    const int m0   = blockIdx.x * 16;
    const int n0   = blockIdx.y * 64;

    v8f acc[4] = {};
    const float* arow = X + (size_t)(m0 + ln) * D_;

    for (int kk = 0; kk < D_; kk += 32) {
        Frag a;
#pragma unroll
        for (int e = 0; e < 8; ++e) {
            a.v[e]     = (__bf16)arow[kk + khB + e];        // K = kk+khB+e
            a.v[e + 8] = (__bf16)arow[kk + khB + 8 + e];    // K = kk+khB+8+e
        }
#pragma unroll
        for (int j = 0; j < 4; ++j) {
            const float* wrow = W + (size_t)(n0 + j * 16 + ln) * D_ + kk + khB;
            Frag bf;
#pragma unroll
            for (int e = 0; e < 16; ++e) bf.v[e] = (__bf16)wrow[e];
            acc[j] = __builtin_amdgcn_wmma_f32_16x16x32_bf16(
                false, a.v, false, bf.v, (short)0, acc[j], false, false);
        }
    }

#pragma unroll
    for (int j = 0; j < 4; ++j) {
        const int   n  = n0 + j * 16 + ln;
        const float bn = bias[n];
#pragma unroll
        for (int r = 0; r < 8; ++r) {
            const int row = m0 + r + 8 * g;               // C layout: VGPR r -> row r+8g
            float val = acc[j][r] + bn;
            val = val > 0.f ? val : 0.f;
            Y[(size_t)row * HID_ + n] = (__bf16)(val * scale);
        }
    }
}

// ---------------------------------------------------------------------------
// Kernel 2: Vt[b][h][vd][s] = (bf16) V[b][s][h*64+vd]
// LDS-tiled 64x64 transpose: coalesced float4 reads, coalesced 8B bf16 writes.
// LDS row stride 66 elems -> 33 dwords, gcd(33,64)=1 -> conflict-free columns.
// ---------------------------------------------------------------------------
__global__ __launch_bounds__(256) void pack_v_kernel(
    const float* __restrict__ V, __hip_bfloat16* __restrict__ Vto)
{
    __bf16* Vt = (__bf16*)Vto;
    __shared__ __bf16 tile[64][66];

    const int tid = threadIdx.x;
    const int s0  = blockIdx.x * 64;
    const int h   = blockIdx.y;
    const int b   = blockIdx.z;

    const float* src = V + ((size_t)b * S_ + s0) * DV_ + h * HDI;
    const int jr = (tid & 15) * 4;     // vd within row (float4)
    const int ir = tid >> 4;           // s row, 16 rows per pass
#pragma unroll
    for (int p = 0; p < 4; ++p) {
        const int i = ir + p * 16;
        const float4 f = *(const float4*)(src + (size_t)i * DV_ + jr);
        tile[i][jr + 0] = (__bf16)f.x;
        tile[i][jr + 1] = (__bf16)f.y;
        tile[i][jr + 2] = (__bf16)f.z;
        tile[i][jr + 3] = (__bf16)f.w;
    }
    __syncthreads();

    __bf16* dst = Vt + ((size_t)b * H_ + h) * (size_t)HDI * S_ + s0;
    const int sw  = (tid & 15) * 4;    // s within row (4 x bf16 = 8B)
    const int vdr = tid >> 4;          // vd row, 16 rows per pass
#pragma unroll
    for (int p = 0; p < 4; ++p) {
        const int vd = vdr + p * 16;
        union { __bf16 h4[4]; uint2 u; } o;
        o.h4[0] = tile[sw + 0][vd];
        o.h4[1] = tile[sw + 1][vd];
        o.h4[2] = tile[sw + 2][vd];
        o.h4[3] = tile[sw + 3][vd];
        *(uint2*)(dst + (size_t)vd * S_ + sw) = o.u;
    }
}

// ---------------------------------------------------------------------------
// Kernel 3: flash attention, one wave per (b, h, 32-row q tile).
// 2 M-tiles per wave double the WMMA work per streamed K/V fragment.
// ---------------------------------------------------------------------------
__global__ __launch_bounds__(32) void attn_kernel(
    const __hip_bfloat16* __restrict__ Qo,   // [B, T, HID] bf16 (relu+scaled)
    const __hip_bfloat16* __restrict__ Ko,   // [B, S, HID] bf16 (relu)
    const __hip_bfloat16* __restrict__ Vto,  // [B, H, 64, S] bf16
    const unsigned char*  __restrict__ mask, // [B, S] (all-false here)
    float* __restrict__ O)                   // [B, T, DV] f32
{
    const __bf16* Q  = (const __bf16*)Qo;
    const __bf16* K  = (const __bf16*)Ko;
    const __bf16* Vt = (const __bf16*)Vto;

    __shared__ __bf16 Pl[32 * 40];      // 32x32 P tile, stride 40 (16B-aligned rows)

    const int lane = threadIdx.x;
    const int g    = lane >> 4;
    const int ln   = lane & 15;
    const int khA  = g ? 8 : 0;
    const int khB  = g ? 16 : 0;
    const int t0   = blockIdx.x * 32;
    const int h    = blockIdx.y;
    const int b    = blockIdx.z;

    // Q A-fragments: 2 M-tiles x 2 K-steps (hd 0..31, 32..63), kept resident.
    Frag qa[2][2];
#pragma unroll
    for (int mt = 0; mt < 2; ++mt) {
        const __bf16* qrow =
            Q + ((size_t)b * T_ + t0 + mt * 16 + ln) * HID_ + h * HDI;
#pragma unroll
        for (int ks = 0; ks < 2; ++ks) {
            qa[mt][ks].u[0] = *(const uint4*)(qrow + ks * 32 + khA);
            qa[mt][ks].u[1] = *(const uint4*)(qrow + ks * 32 + 16 + khA);
        }
    }

    float mrun[2][8], lrun[2][8];
    v8f acc[2][4] = {};
#pragma unroll
    for (int mt = 0; mt < 2; ++mt)
#pragma unroll
        for (int r = 0; r < 8; ++r) { mrun[mt][r] = -1e30f; lrun[mt][r] = 0.f; }

    const __bf16* kbase = K + (size_t)b * S_ * HID_ + h * HDI;
    const __bf16* vbase = Vt + ((size_t)b * H_ + h) * (size_t)HDI * S_;
    const unsigned char* mbase = mask + (size_t)b * S_;

    for (int s0 = 0; s0 < S_; s0 += 32) {
        // Prefetch next chunk's K rows (32) and V rows (64) -> global_prefetch_b8
        if (s0 + 32 < S_) {
            __builtin_prefetch(kbase + (size_t)(s0 + 32 + lane) * HID_, 0, 1);
            __builtin_prefetch(vbase + (size_t)lane * S_ + s0 + 32, 0, 1);
            __builtin_prefetch(vbase + (size_t)(lane + 32) * S_ + s0 + 32, 0, 1);
        }

        // ---- scores: 2 M-tiles x 2 S-tiles; K fragments reused across M ----
        v8f  c[2][2];
        bool mk[2];
#pragma unroll
        for (int st = 0; st < 2; ++st) {
            const int s = s0 + st * 16 + ln;          // B column = key row
            const __bf16* krow = kbase + (size_t)s * HID_;
            Frag kb0, kb1;                             // K = ks*32 + khA + [0..7], ks*32 + 16 + khA + [0..7]
            kb0.u[0] = *(const uint4*)(krow + khA);
            kb0.u[1] = *(const uint4*)(krow + 16 + khA);
            kb1.u[0] = *(const uint4*)(krow + 32 + khA);
            kb1.u[1] = *(const uint4*)(krow + 32 + 16 + khA);
            mk[st] = mbase[s] != 0;
#pragma unroll
            for (int mt = 0; mt < 2; ++mt) {
                v8f ct = {};
                ct = __builtin_amdgcn_wmma_f32_16x16x32_bf16(
                    false, qa[mt][0].v, false, kb0.v, (short)0, ct, false, false);
                ct = __builtin_amdgcn_wmma_f32_16x16x32_bf16(
                    false, qa[mt][1].v, false, kb1.v, (short)0, ct, false, false);
                c[mt][st] = ct;
            }
        }
#pragma unroll
        for (int st = 0; st < 2; ++st)
            if (mk[st]) {
#pragma unroll
                for (int mt = 0; mt < 2; ++mt)
#pragma unroll
                    for (int r = 0; r < 8; ++r) c[mt][st][r] = -1e30f;
            }

        // ---- online softmax per M-tile (row r+8g lives in VGPR r) ----
#pragma unroll
        for (int mt = 0; mt < 2; ++mt) {
            float p0[8], p1[8], alpha[8];
#pragma unroll
            for (int r = 0; r < 8; ++r) {
                float t = fmaxf(c[mt][0][r], c[mt][1][r]);
#pragma unroll
                for (int off = 1; off < 16; off <<= 1)
                    t = fmaxf(t, __shfl_xor(t, off, 32));
                const float mn = fmaxf(mrun[mt][r], t);
                alpha[r] = exp2f((mrun[mt][r] - mn) * 1.44269504f);
                p0[r]    = exp2f((c[mt][0][r] - mn) * 1.44269504f);
                p1[r]    = exp2f((c[mt][1][r] - mn) * 1.44269504f);
                float rs = p0[r] + p1[r];
#pragma unroll
                for (int off = 1; off < 16; off <<= 1)
                    rs += __shfl_xor(rs, off, 32);
                lrun[mt][r] = lrun[mt][r] * alpha[r] + rs;
                mrun[mt][r] = mn;
            }
            // P (C layout) -> LDS, and rescale this tile's accumulators
#pragma unroll
            for (int r = 0; r < 8; ++r) {
                Pl[(mt * 16 + r + 8 * g) * 40 + ln]      = (__bf16)p0[r];
                Pl[(mt * 16 + r + 8 * g) * 40 + 16 + ln] = (__bf16)p1[r];
            }
#pragma unroll
            for (int j = 0; j < 4; ++j)
#pragma unroll
                for (int r = 0; r < 8; ++r) acc[mt][j][r] *= alpha[r];
        }
        __syncthreads();

        // P A-fragments (one per M-tile)
        Frag pa[2];
#pragma unroll
        for (int mt = 0; mt < 2; ++mt) {
            pa[mt].u[0] = *(const uint4*)(&Pl[(mt * 16 + ln) * 40 + khA]);
            pa[mt].u[1] = *(const uint4*)(&Pl[(mt * 16 + ln) * 40 + 16 + khA]);
        }

        // ---- acc += P x V : 4 vd tiles, V fragment reused across M ----
#pragma unroll
        for (int j = 0; j < 4; ++j) {
            const __bf16* vrow = vbase + (size_t)(j * 16 + ln) * S_ + s0 + khA;
            Frag vb;
            vb.u[0] = *(const uint4*)(vrow);
            vb.u[1] = *(const uint4*)(vrow + 16);
#pragma unroll
            for (int mt = 0; mt < 2; ++mt)
                acc[mt][j] = __builtin_amdgcn_wmma_f32_16x16x32_bf16(
                    false, pa[mt].v, false, vb.v, (short)0, acc[mt][j], false, false);
        }
        __syncthreads();   // protect Pl against next iteration's writes
    }

    // ---- epilogue ----
#pragma unroll
    for (int mt = 0; mt < 2; ++mt)
#pragma unroll
        for (int r = 0; r < 8; ++r) {
            const float inv = 1.0f / lrun[mt][r];
            const int row   = t0 + mt * 16 + r + 8 * g;
            float* orow = O + ((size_t)b * T_ + row) * DV_ + h * HDI;
#pragma unroll
            for (int j = 0; j < 4; ++j)
                orow[j * 16 + ln] = acc[mt][j][r] * inv;
        }
}

// ---------------------------------------------------------------------------
extern "C" void kernel_launch(void* const* d_in, const int* in_sizes, int n_in,
                              void* d_out, int out_size, void* d_ws, size_t ws_size,
                              hipStream_t stream)
{
    const float* query = (const float*)d_in[0];
    const float* key   = (const float*)d_in[1];
    const float* value = (const float*)d_in[2];
    const unsigned char* mask = (const unsigned char*)d_in[3];
    const float* Wq = (const float*)d_in[4];
    const float* bq = (const float*)d_in[5];
    const float* Wk = (const float*)d_in[6];
    const float* bk = (const float*)d_in[7];

    char* ws = (char*)d_ws;
    const size_t QB = (size_t)B_ * T_ * HID_ * sizeof(__hip_bfloat16);
    const size_t KB = (size_t)B_ * S_ * HID_ * sizeof(__hip_bfloat16);
    __hip_bfloat16* qp = (__hip_bfloat16*)(ws);
    __hip_bfloat16* kp = (__hip_bfloat16*)(ws + QB);
    __hip_bfloat16* vt = (__hip_bfloat16*)(ws + QB + KB);

    const float scaling = 0.125f;   // HD^-0.5 = 64^-0.5

    dim3 pgrid(B_ * T_ / 16, HID_ / 64);
    proj_relu_kernel<<<pgrid, 32, 0, stream>>>(query, Wq, bq, qp, scaling);
    proj_relu_kernel<<<pgrid, 32, 0, stream>>>(key,   Wk, bk, kp, 1.0f);

    pack_v_kernel<<<dim3(S_ / 64, H_, B_), 256, 0, stream>>>(value, vt);

    attn_kernel<<<dim3(T_ / 32, H_, B_), 32, 0, stream>>>(qp, kp, vt, mask,
                                                          (float*)d_out);
}
